// Attention_9423158248136
// MI455X (gfx1250) — matrix operand order repaired and checked by reading, not compiled, not run on hardware
//
#include <hip/hip_runtime.h>

typedef __attribute__((ext_vector_type(16))) _Float16 v16h;
typedef __attribute__((ext_vector_type(8)))  _Float16 v8h;
typedef __attribute__((ext_vector_type(8)))  float    v8f;
typedef __attribute__((ext_vector_type(4)))  int      v4i;

#define B_  4
#define N_  2048
#define D_  512
#define H_  8
#define DH_ 64

#if defined(__has_builtin)
#if __has_builtin(__builtin_amdgcn_global_load_async_to_lds_b128) && \
    __has_builtin(__builtin_amdgcn_s_wait_asynccnt)
#define USE_ASYNC_LDS 1
#endif
#endif

#define AS1 __attribute__((address_space(1)))
#define AS3 __attribute__((address_space(3)))

__device__ __forceinline__ v8f wmma_f16(v16h a, v16h b, v8f c) {
  // D = A*B + C, 16x16x32, f16 in / f32 acc
  return __builtin_amdgcn_wmma_f32_16x16x32_f16(
      false, a, false, b, (short)0, c, false, false);
}

// Unused: in AMD table order, element i of lane-half hf holds k = kk*32 + swap34(hf*16 + i).
// Storing data with bits 3 and 4 of the k-index swapped makes the fragment a
// single contiguous 16-half (32 B) load at offset kk*32 + hf*16.
__device__ __forceinline__ int swap34(int c) {
  return (c & ~24) | ((c & 8) << 1) | ((c & 16) >> 1);
}

// ---------------------------------------------------------------------------
// Kernel 0: weights -> f16, transposed so B-fragments are contiguous v16h
// ---------------------------------------------------------------------------
__global__ __launch_bounds__(256) void wconv_kernel(
    const float* __restrict__ w_qkv, const float* __restrict__ w_out,
    _Float16* __restrict__ wqkvt, _Float16* __restrict__ woutt)
{
  const int idx = blockIdx.x * 256 + threadIdx.x;
  if (idx < D_ * 3 * D_) {          // w_qkv [512][1536] -> wqkvt [1536][512]
    const int k = idx / (3 * D_), c = idx % (3 * D_);
    wqkvt[(size_t)c * D_ + k] = (_Float16)w_qkv[idx];
  }
  if (idx < D_ * D_) {              // w_out [512][512] -> woutt [512][512]^T
    const int k = idx / D_, c = idx % D_;
    woutt[(size_t)c * D_ + k] = (_Float16)w_out[idx];
  }
}

// ---------------------------------------------------------------------------
// Kernel 1: QKV projection  x[8192,512] @ w_qkv -> f16 Q/K/Vt
// ---------------------------------------------------------------------------
__global__ __launch_bounds__(128) void qkv_gemm_kernel(
    const float* __restrict__ x, const _Float16* __restrict__ wqkvt,
    _Float16* __restrict__ qf, _Float16* __restrict__ kf,
    _Float16* __restrict__ vt)
{
  const int wave = threadIdx.x >> 5;
  const int lane = threadIdx.x & 31;
  const int hf   = lane >> 4;
  const int ln   = lane & 15;

  const int mtile = blockIdx.x;              // 512 tiles over 8192 rows
  const int ntile = blockIdx.y * 4 + wave;   // 96 tiles over 1536 cols
  const int row   = mtile * 16 + ln;
  const int col   = ntile * 16 + ln;

  const float*    ap = x + (size_t)row * D_ + hf * 16;
  const _Float16* bp = wqkvt + (size_t)col * D_ + hf * 16;

  v8f acc = {};
  for (int kk = 0; kk < D_; kk += 32) {
    v8f lo = *(const v8f*)(ap + kk);         // k = kk + hf*16 + 0..7
    v8f hi = *(const v8f*)(ap + kk + 8);     // k = kk + hf*16 + 8..15
    v16h a;
#pragma unroll
    for (int i = 0; i < 8; ++i) { a[i] = (_Float16)lo[i]; a[8 + i] = (_Float16)hi[i]; }
    v16h b = *(const v16h*)(bp + kk);        // contiguous: k = kk + hf*16 + i
    acc = wmma_f16(a, b, acc);
  }

  const int which = col / D_;                // 0=Q 1=K 2=V  (uniform per tile)
  const int head  = (col % D_) / DH_;        // uniform per tile
  const int d     = col % DH_;
  const int dq    = d;                       // Q stored unpermuted
#pragma unroll
  for (int j = 0; j < 8; ++j) {
    const int m  = mtile * 16 + j + hf * 8;  // global row in [0, B*N)
    const int bb = m / N_;
    const int n  = m % N_;
    const float v = acc[j];
    if (which == 0)
      qf[((size_t)(bb * H_ + head) * N_ + n) * DH_ + dq] = (_Float16)(v * 0.125f);
    else if (which == 1)
      kf[((size_t)(bb * H_ + head) * N_ + n) * DH_ + d] = (_Float16)v;
    else
      vt[((size_t)(bb * H_ + head) * DH_ + d) * N_ + n] = (_Float16)v;
  }
}

// ---------------------------------------------------------------------------
// Kernel 2: flash attention; block = one (b,h) x 128 query rows, 8 waves.
// K/V tiles staged cooperatively in LDS (async-to-LDS when available).
// ---------------------------------------------------------------------------
__global__ __launch_bounds__(256) void attn_kernel(
    const _Float16* __restrict__ qf, const _Float16* __restrict__ kf,
    const _Float16* __restrict__ vt, const float* __restrict__ bias,
    _Float16* __restrict__ ao)
{
  __shared__ alignas(32) _Float16 kbuf[32][64];   // [key][d]        4 KB
  __shared__ alignas(32) _Float16 vbuf[64][32];   // [d][key]        4 KB
  __shared__ alignas(32) _Float16 pbuf[8][16][32];// per-wave P      8 KB

  const int tid  = threadIdx.x;
  const int wave = tid >> 5;
  const int lane = tid & 31;
  const int hf   = lane >> 4;
  const int ln   = lane & 15;

  const int bh = blockIdx.x;               // b*H + h
  const int h  = bh & (H_ - 1);
  const int bb = bh / H_;
  const int q0 = blockIdx.y * 128 + wave * 16;

  const _Float16* Qp = qf + (size_t)bh * N_ * DH_;
  const _Float16* Kp = kf + (size_t)bh * N_ * DH_;
  const _Float16* Vp = vt + (size_t)bh * DH_ * N_;
  const float*    Bp = bias + (size_t)h * N_ * N_;

  // Q fragments: unpermuted Q, contiguous k = kk*32 + hf*16 + i as for K; reused for all steps
  v16h qa[2];
#pragma unroll
  for (int kk = 0; kk < 2; ++kk)
    qa[kk] = *(const v16h*)(Qp + (size_t)(q0 + ln) * DH_ + kk * 32 + hf * 16);

  v8f o[4];
#pragma unroll
  for (int t = 0; t < 4; ++t)
#pragma unroll
    for (int j = 0; j < 8; ++j) o[t][j] = 0.0f;
  float rmax[8], rsum[8];
#pragma unroll
  for (int j = 0; j < 8; ++j) { rmax[j] = -3.0e38f; rsum[j] = 0.0f; }

  // staging assignments (256 threads x 16B cover each 4 KB tile)
  const int kr = tid >> 3, ko = (tid & 7) * 8;   // kbuf[32][64]
  const int vr = tid >> 2, vo = (tid & 3) * 8;   // vbuf[64][32]

  for (int j0 = 0; j0 < N_; j0 += 32) {
    __syncthreads();   // previous-iteration readers of kbuf/vbuf are done
#ifdef USE_ASYNC_LDS
    __builtin_amdgcn_global_load_async_to_lds_b128(
        (AS1 v4i*)(Kp + (size_t)(j0 + kr) * DH_ + ko),
        (AS3 v4i*)&kbuf[kr][ko], 0, 0);
    __builtin_amdgcn_global_load_async_to_lds_b128(
        (AS1 v4i*)(Vp + (size_t)vr * N_ + j0 + vo),
        (AS3 v4i*)&vbuf[vr][vo], 0, 0);
    __builtin_amdgcn_s_wait_asynccnt(0);
#else
    *(v8h*)&kbuf[kr][ko] = *(const v8h*)(Kp + (size_t)(j0 + kr) * DH_ + ko);
    *(v8h*)&vbuf[vr][vo] = *(const v8h*)(Vp + (size_t)vr * N_ + j0 + vo);
#endif
    if (j0 + 32 < N_)  // hint: next bias tile (the only HBM-bound stream)
      __builtin_prefetch(Bp + (size_t)(q0 + ln) * N_ + j0 + 32, 0, 1);
    __syncthreads();

    // ---- S = Q K^T : 16x32 score tile (two 16x16 sub-tiles) ----
    v8f s[2];
#pragma unroll
    for (int t = 0; t < 2; ++t) {
      v8f c = {};
#pragma unroll
      for (int kk = 0; kk < 2; ++kk) {
        v16h kb = *(const v16h*)&kbuf[t * 16 + ln][kk * 32 + hf * 16];
        c = wmma_f16(qa[kk], kb, c);
      }
      const int jcol = j0 + t * 16 + ln;
#pragma unroll
      for (int j = 0; j < 8; ++j)
        c[j] += Bp[(size_t)(q0 + j + hf * 8) * N_ + jcol];
      s[t] = c;
    }

    // ---- online softmax (row reductions across 16-lane groups) ----
#pragma unroll
    for (int j = 0; j < 8; ++j) {
      float m = fmaxf(s[0][j], s[1][j]);
#pragma unroll
      for (int off = 1; off < 16; off <<= 1)
        m = fmaxf(m, __shfl_xor(m, off, 32));
      const float nm = fmaxf(rmax[j], m);
      const float sc = __expf(rmax[j] - nm);
      rmax[j] = nm;
      o[0][j] *= sc; o[1][j] *= sc; o[2][j] *= sc; o[3][j] *= sc;
      const float e0 = __expf(s[0][j] - nm);
      const float e1 = __expf(s[1][j] - nm);
      s[0][j] = e0; s[1][j] = e1;
      float ssum = e0 + e1;
#pragma unroll
      for (int off = 1; off < 16; off <<= 1)
        ssum += __shfl_xor(ssum, off, 32);
      rsum[j] = rsum[j] * sc + ssum;
    }

    // ---- P: C-layout -> A-layout in per-wave LDS tile ----
#pragma unroll
    for (int t = 0; t < 2; ++t)
#pragma unroll
      for (int j = 0; j < 8; ++j)
        pbuf[wave][j + hf * 8][t * 16 + ln] = (_Float16)s[t][j];
    __syncthreads();

    v16h pa = *(const v16h*)&pbuf[wave][ln][hf * 16];

    // ---- O += P @ V  (4 d-subtiles of 16) ----
#pragma unroll
    for (int t = 0; t < 4; ++t) {
      v16h vb = *(const v16h*)&vbuf[t * 16 + ln][hf * 16];
      o[t] = wmma_f16(pa, vb, o[t]);
    }
  }

  // ---- normalize, store AO f16 in [B,N,512] ----
#pragma unroll
  for (int t = 0; t < 4; ++t)
#pragma unroll
    for (int j = 0; j < 8; ++j) {
      const int qr = q0 + j + hf * 8;
      const int cl = h * DH_ + t * 16 + ln;
      ao[((size_t)bb * N_ + qr) * D_ + cl] = (_Float16)(o[t][j] / rsum[j]);
    }
}

// ---------------------------------------------------------------------------
// Kernel 3: output projection  AO[8192,512] @ w_out -> out f32
// ---------------------------------------------------------------------------
__global__ __launch_bounds__(128) void out_gemm_kernel(
    const _Float16* __restrict__ ao, const _Float16* __restrict__ woutt,
    float* __restrict__ out)
{
  const int wave = threadIdx.x >> 5;
  const int lane = threadIdx.x & 31;
  const int hf   = lane >> 4;
  const int ln   = lane & 15;

  const int mtile = blockIdx.x;              // 512
  const int ntile = blockIdx.y * 4 + wave;   // 32
  const int row   = mtile * 16 + ln;
  const int col   = ntile * 16 + ln;

  const _Float16* app = ao + (size_t)row * D_ + hf * 16;     // unpermuted A
  const _Float16* bpp = woutt + (size_t)col * D_ + hf * 16;  // transposed B

  v8f acc = {};
  for (int kk = 0; kk < D_; kk += 32) {
    v16h a = *(const v16h*)(app + kk);
    v16h b = *(const v16h*)(bpp + kk);
    acc = wmma_f16(a, b, acc);
  }
#pragma unroll
  for (int j = 0; j < 8; ++j) {
    const int m = mtile * 16 + j + hf * 8;
    out[(size_t)m * D_ + col] = acc[j];
  }
}

// ---------------------------------------------------------------------------
extern "C" void kernel_launch(void* const* d_in, const int* in_sizes, int n_in,
                              void* d_out, int out_size, void* d_ws,
                              size_t ws_size, hipStream_t stream) {
  (void)in_sizes; (void)n_in; (void)out_size; (void)ws_size;
  const float* x     = (const float*)d_in[0];
  const float* bias  = (const float*)d_in[1];
  const float* w_qkv = (const float*)d_in[2];
  const float* w_out = (const float*)d_in[3];
  float* out = (float*)d_out;

  const size_t per = (size_t)B_ * H_ * N_ * DH_;   // 4,194,304 f16 elems
  _Float16* qf    = (_Float16*)d_ws;               // 8 MB
  _Float16* kf    = qf + per;                      // 8 MB
  _Float16* vt    = kf + per;                      // 8 MB (transposed V)
  _Float16* aoBuf = vt + per;                      // 8 MB
  _Float16* wqkvt = aoBuf + per;                   // 1.5 MB
  _Float16* woutt = wqkvt + (size_t)3 * D_ * D_;   // 0.5 MB

  dim3 g0((D_ * 3 * D_ + 255) / 256), b0(256);
  wconv_kernel<<<g0, b0, 0, stream>>>(w_qkv, w_out, wqkvt, woutt);

  dim3 g1(512, 24), b1(128);
  qkv_gemm_kernel<<<g1, b1, 0, stream>>>(x, wqkvt, qf, kf, vt);

  dim3 g2(B_ * H_, N_ / 128), b2(256);
  attn_kernel<<<g2, b2, 0, stream>>>(qf, kf, vt, bias, aoBuf);

  dim3 g3(512, 8), b3(128);
  out_gemm_kernel<<<g3, b3, 0, stream>>>(aoBuf, woutt, out);
}
